// RiemannianFlowMatcher_7275674599709
// MI455X (gfx1250) — matrix operand order repaired and checked by reading, not compiled, not run on hardware
//
#include <hip/hip_runtime.h>
#include <hip/hip_bf16.h>
#include <math.h>

// ---------------------------------------------------------------------------
// RiemannianFlowMatcher forward, CDNA5 (gfx1250) WMMA implementation.
// H=256, L=6, B=4, N=128, NE=100, NP=5, NF=128
// Dominant work: per-layer edge GEMM (B*N*N, H) @ (H, H)  ~8.6 GFLOP/layer,
// fused so the (B,N,N,H) tensor never touches HBM.  eW2 is pre-converted to
// f16 once and staged into LDS with global_load_async_to_lds_b128 (ASYNCcnt).
// ---------------------------------------------------------------------------

typedef __attribute__((ext_vector_type(16))) _Float16 h16x16;
typedef __attribute__((ext_vector_type(8)))  float    f32x8;

#define TWO_PI_F 6.28318530717958647692f

// fast silu: x * rcp(1 + exp(-x))  (v_exp_f32 + v_rcp_f32, no div_fixup chain)
__device__ __forceinline__ float silu_f(float x) {
  return x * __builtin_amdgcn_rcpf(1.0f + __expf(-x));
}

// A-fragment: 16x32 f16, lane<16 holds row M=lane with K {0..7,16..23},
// lane>=16 holds row M=lane-16 with K {8..15,24..31}.  `row` -> 32 contiguous f16.
__device__ __forceinline__ h16x16 load_a_frag(const _Float16* row, int half) {
  h16x16 a;
  const int kb1 = half * 8;
  const int kb2 = 16 + half * 8;
#pragma unroll
  for (int q = 0; q < 8; ++q) a[q] = row[kb1 + q];
#pragma unroll
  for (int q = 0; q < 8; ++q) a[8 + q] = row[kb2 + q];
  return a;
}

// B-fragment: 32x16 f16 staged [n][k] in LDS; lane<16 holds col N=lane K {0..7,16..23},
// lane>=16 holds col N=lane-16 K {8..15,24..31}.  `col` -> 32 contiguous f16.
__device__ __forceinline__ h16x16 load_b_frag(const _Float16* col, int half) {
  h16x16 b;
  const _Float16* p = col + half * 8;
#pragma unroll
  for (int q = 0; q < 8; ++q) b[q] = p[q];
#pragma unroll
  for (int q = 0; q < 8; ++q) b[8 + q] = p[16 + q];
  return b;
}

// ---------------------------------------------------------------------------
// Generic 256-wide WMMA GEMM:  C[M,256] = op(A[M,256] @ W[256,256] (+ A2@W2)
//                                           + bias) (+ Cin)
// Block = 256 threads (8 wave32), tile M=64, full N=256, K sliced by 32.
// ---------------------------------------------------------------------------
template<bool DUAL, bool SILU, bool ACCUM_C, bool BIAS>
__global__ __launch_bounds__(256)
void gemm256_wmma(const float* __restrict__ A,  const float* __restrict__ W,
                  const float* __restrict__ A2, const float* __restrict__ W2,
                  const float* __restrict__ bias, const float* __restrict__ Cin,
                  float* __restrict__ C) {
  __shared__ __align__(64) _Float16 Asl[64][32];
  __shared__ __align__(64) _Float16 Bsl[256][32];

  const int tid  = threadIdx.x;
  const int wave = tid >> 5;
  const int lane = tid & 31;
  const int half = lane >> 4;
  const int mr   = lane & 15;
  const int m0   = blockIdx.x * 64;
  const int n0   = wave * 32;

  const f32x8 zacc = {0.f, 0.f, 0.f, 0.f, 0.f, 0.f, 0.f, 0.f};
  f32x8 acc[4][2];
#pragma unroll
  for (int mt = 0; mt < 4; ++mt)
#pragma unroll
    for (int nt = 0; nt < 2; ++nt) acc[mt][nt] = zacc;

  const int npass = DUAL ? 2 : 1;
  for (int pass = 0; pass < npass; ++pass) {
    const float* Ap = (DUAL && pass) ? A2 : A;
    const float* Wp = (DUAL && pass) ? W2 : W;
    for (int ks = 0; ks < 8; ++ks) {
      const int k0 = ks * 32;
      {  // stage A tile 64x32 (f32 -> f16)
        const int m  = tid >> 2;
        const int kq = (tid & 3) * 8;
        const float* src = Ap + (m0 + m) * 256 + k0 + kq;
#pragma unroll
        for (int q = 0; q < 8; ++q) Asl[m][kq + q] = (_Float16)src[q];
      }
      {  // stage W K-slice transposed -> Bsl[n][k]
        const int n = tid;
#pragma unroll
        for (int k = 0; k < 32; ++k)
          Bsl[n][k] = (_Float16)Wp[(k0 + k) * 256 + n];
      }
      __syncthreads();
      h16x16 bf[2];
#pragma unroll
      for (int nt = 0; nt < 2; ++nt)
        bf[nt] = load_b_frag(&Bsl[n0 + nt * 16 + mr][0], half);
#pragma unroll
      for (int mt = 0; mt < 4; ++mt) {
        h16x16 af = load_a_frag(&Asl[mt * 16 + mr][0], half);
#pragma unroll
        for (int nt = 0; nt < 2; ++nt)
          acc[mt][nt] = __builtin_amdgcn_wmma_f32_16x16x32_f16(
              false, af, false, bf[nt], (short)0, acc[mt][nt], false, false);
      }
      __syncthreads();
    }
  }

  // epilogue: D layout — VGPR r: lanes 0-15 M=r, lanes 16-31 M=8+r; N=lane&15
#pragma unroll
  for (int nt = 0; nt < 2; ++nt) {
    const int n  = n0 + nt * 16 + mr;
    const float bv = BIAS ? bias[n] : 0.0f;
#pragma unroll
    for (int mt = 0; mt < 4; ++mt) {
#pragma unroll
      for (int r = 0; r < 8; ++r) {
        const int m = m0 + mt * 16 + half * 8 + r;
        float v = acc[mt][nt][r] + bv;
        if (SILU) v = silu_f(v);
        if (ACCUM_C) v += Cin[m * 256 + n];
        C[m * 256 + n] = v;
      }
    }
  }
}

// ---------------------------------------------------------------------------
// eW2 pre-transpose + f32->f16 conversion (once per launch, all 6 layers):
//   W2T[l][n][k] = eW2[l][k][n]
// ---------------------------------------------------------------------------
__global__ __launch_bounds__(256)
void cvt_w2t(const float* __restrict__ eW2, _Float16* __restrict__ W2T) {
  const int l = blockIdx.x >> 8;
  const int n = blockIdx.x & 255;
  const int k = threadIdx.x;
  W2T[(l * 256 + n) * 256 + k] = (_Float16)eW2[(l * 256 + k) * 256 + n];
}

// ---------------------------------------------------------------------------
// Fused edge kernel: one workgroup per (b,i).
//   e[j,k] = silu(hi[b,i,k] + hj[b,j,k] + dist[b,i,j]*w1d[k] + b1[k])   (LDS f16)
//   msg[b,i,n] = sum_j silu( (e @ eW2)[j,n] + b2[n] )
// M=128 (j), K=256 sliced by 32, N=256 (8 waves x 32 cols). e never hits HBM.
// W2T is f16 [n][k]; staged LDS-direct via global_load_async_to_lds_b128.
// ---------------------------------------------------------------------------
__global__ __launch_bounds__(256)
void edge_msg_wmma(const float* __restrict__ hi_, const float* __restrict__ hj_,
                   const float* __restrict__ dist, const float* __restrict__ w1d,
                   const float* __restrict__ b1,
                   const _Float16* __restrict__ W2T,
                   const float* __restrict__ b2, float* __restrict__ msg) {
  __shared__ __align__(64) _Float16 Ae[128][32];
  __shared__ __align__(64) _Float16 Be[256][32];
  __shared__ float hi_s[256], w1d_s[256], b1_s[256];
  __shared__ float dist_s[128];

  const int tid  = threadIdx.x;
  const int wave = tid >> 5;
  const int lane = tid & 31;
  const int half = lane >> 4;
  const int mr   = lane & 15;
  const int bi   = blockIdx.x;          // b*128 + i
  const int bbase = (bi >> 7) * 128;    // first atom row of this batch
  const int n0   = wave * 32;

  hi_s[tid]  = hi_[bi * 256 + tid];
  w1d_s[tid] = w1d[tid];
  b1_s[tid]  = b1[tid];
  if (tid < 128) dist_s[tid] = dist[bi * 128 + tid];
  __syncthreads();

  const unsigned long long w2t_base = (unsigned long long)W2T;
  const unsigned lds_row = (unsigned)(size_t)(&Be[tid][0]);  // LDS byte offset

  const f32x8 zacc = {0.f, 0.f, 0.f, 0.f, 0.f, 0.f, 0.f, 0.f};
  f32x8 acc[8][2];
#pragma unroll
  for (int jt = 0; jt < 8; ++jt)
#pragma unroll
    for (int nt = 0; nt < 2; ++nt) acc[jt][nt] = zacc;

  for (int ks = 0; ks < 8; ++ks) {
    const int k0 = ks * 32;
    {  // async-stage eW2 f16 K-slice: Be[n][0..31] <- W2T[n*256 + k0 .. +31]
      const unsigned gbyte = (unsigned)(tid * 512 + k0 * 2);  // bytes into W2T
#pragma unroll
      for (int q = 0; q < 4; ++q) {
        const unsigned l_off = lds_row + (unsigned)(q * 16);
        const unsigned g_off = gbyte + (unsigned)(q * 16);
        asm volatile("global_load_async_to_lds_b128 %0, %1, %2"
                     :: "v"(l_off), "v"(g_off), "s"(w2t_base) : "memory");
      }
    }
    {  // build e slice 128x32 directly in LDS (f16)
      const int j  = tid >> 1;
      const int kq = (tid & 1) * 16;
      const float* hjrow = hj_ + (bbase + j) * 256 + k0 + kq;
      const float dj = dist_s[j];
#pragma unroll
      for (int q = 0; q < 16; ++q) {
        const int k = k0 + kq + q;
        const float v = hi_s[k] + hjrow[q] + dj * w1d_s[k] + b1_s[k];
        Ae[j][kq + q] = (_Float16)silu_f(v);
      }
    }
    asm volatile("s_wait_asynccnt 0x0" ::: "memory");
    __syncthreads();
    h16x16 bf[2];
#pragma unroll
    for (int nt = 0; nt < 2; ++nt)
      bf[nt] = load_b_frag(&Be[n0 + nt * 16 + mr][0], half);
#pragma unroll
    for (int jt = 0; jt < 8; ++jt) {
      h16x16 af = load_a_frag(&Ae[jt * 16 + mr][0], half);
#pragma unroll
      for (int nt = 0; nt < 2; ++nt)
        acc[jt][nt] = __builtin_amdgcn_wmma_f32_16x16x32_f16(
            false, af, false, bf[nt], (short)0, acc[jt][nt], false, false);
    }
    __syncthreads();
  }

  // epilogue: +b2, silu, reduce over all 128 j rows; finish with xor-16 shuffle
#pragma unroll
  for (int nt = 0; nt < 2; ++nt) {
    const int n   = n0 + nt * 16 + mr;
    const float bv = b2[n];
    float s = 0.0f;
#pragma unroll
    for (int jt = 0; jt < 8; ++jt)
#pragma unroll
      for (int r = 0; r < 8; ++r)
        s += silu_f(acc[jt][nt][r] + bv);
    s += __shfl_xor(s, 16, 32);   // combine the two M-halves (same n)
    if (half == 0) msg[bi * 256 + n] = s;
  }
}

// ---------------------------------------------------------------------------
// Small VALU kernels (negligible FLOPs)
// ---------------------------------------------------------------------------

__global__ __launch_bounds__(256)
void prelude_cond(const float* __restrict__ t_in, const float* __restrict__ props,
                  const float* __restrict__ tW1, const float* __restrict__ tb1,
                  const float* __restrict__ tW2, const float* __restrict__ tb2,
                  const float* __restrict__ pW1, const float* __restrict__ pb1,
                  const float* __restrict__ pW2, const float* __restrict__ pb2,
                  float* __restrict__ cond) {
  __shared__ float ff[256], hid[256];
  const int b = blockIdx.x, t = threadIdx.x;
  const float tt  = fminf(fmaxf(t_in[b], 0.0f), 1.0f);
  const int   fi  = t & 127;
  const float ang = tt * exp2f((float)fi) * TWO_PI_F;
  ff[t] = (t < 128) ? sinf(ang) : cosf(ang);
  __syncthreads();
  float s = tb1[t];
  for (int k = 0; k < 256; ++k) s += ff[k] * tW1[k * 256 + t];
  hid[t] = silu_f(s);
  __syncthreads();
  float o1 = tb2[t];
  for (int k = 0; k < 256; ++k) o1 += hid[k] * tW2[k * 256 + t];
  __syncthreads();
  float s2 = pb1[t];
  for (int k = 0; k < 5; ++k) s2 += props[b * 5 + k] * pW1[k * 256 + t];
  hid[t] = silu_f(s2);
  __syncthreads();
  float o2 = pb2[t];
  for (int k = 0; k < 256; ++k) o2 += hid[k] * pW2[k * 256 + t];
  cond[b * 256 + t] = o1 + o2;
}

__global__ __launch_bounds__(256)
void embed_h(const int* __restrict__ at, const float* __restrict__ fc,
             const float* __restrict__ emb, const float* __restrict__ cW,
             const float* __restrict__ cb, const float* __restrict__ cond,
             float* __restrict__ h) {
  const int idx = blockIdx.x, t = threadIdx.x;
  const int b = idx >> 7;
  const int a = at[idx];
  float v = emb[a * 256 + t] + cb[t] + cond[b * 256 + t];
#pragma unroll
  for (int d = 0; d < 3; ++d) v += fc[idx * 3 + d] * cW[d * 256 + t];
  h[idx * 256 + t] = v;
}

__global__ __launch_bounds__(128)
void dist_pbc(const float* __restrict__ fc, float* __restrict__ dist) {
  const int bi = blockIdx.x, j = threadIdx.x;
  const int bbase = (bi >> 7) * 128;
  float s = 1e-8f;
#pragma unroll
  for (int d = 0; d < 3; ++d) {
    float dx = fc[(bbase + j) * 3 + d] - fc[bi * 3 + d];
    dx -= rintf(dx);  // round-half-even, matches jnp.round
    s += dx * dx;
  }
  dist[bi * 128 + j] = sqrtf(s);
}

__global__ __launch_bounds__(128)
void atom_head(const float* __restrict__ h, const float* __restrict__ aW,
               const float* __restrict__ ab, float* __restrict__ out) {
  const int idx = blockIdx.x, t = threadIdx.x;
  if (t >= 100) return;
  float s = ab[t];
  for (int k = 0; k < 256; ++k) s += h[idx * 256 + k] * aW[k * 100 + t];
  out[idx * 100 + t] = s;
}

__global__ __launch_bounds__(256)
void coord_head(const float* __restrict__ h, const float* __restrict__ W1,
                const float* __restrict__ b1v, const float* __restrict__ W2,
                const float* __restrict__ b2v, float* __restrict__ out) {
  __shared__ float hid[256];
  const int idx = blockIdx.x, t = threadIdx.x;
  float s = b1v[t];
  for (int k = 0; k < 256; ++k) s += h[idx * 256 + k] * W1[k * 256 + t];
  hid[t] = silu_f(s);
  __syncthreads();
  if (t < 3) {
    float s2 = b2v[t];
    for (int k = 0; k < 256; ++k) s2 += hid[k] * W2[k * 3 + t];
    out[idx * 3 + t] = s2;
  }
}

__global__ __launch_bounds__(256)
void lattice_head(const float* __restrict__ h, const float* __restrict__ W1,
                  const float* __restrict__ b1v, const float* __restrict__ W2,
                  const float* __restrict__ b2v, float* __restrict__ out) {
  __shared__ float feat[256], hid[256];
  const int b = blockIdx.x, t = threadIdx.x;
  float s = 0.0f;
  for (int n = 0; n < 128; ++n) s += h[(b * 128 + n) * 256 + t];
  feat[t] = s * (1.0f / 128.0f);
  __syncthreads();
  float s2 = b1v[t];
  for (int k = 0; k < 256; ++k) s2 += feat[k] * W1[k * 256 + t];
  hid[t] = silu_f(s2);
  __syncthreads();
  if (t < 6) {
    float s3 = b2v[t];
    for (int k = 0; k < 256; ++k) s3 += hid[k] * W2[k * 6 + t];
    out[b * 6 + t] = s3;
  }
}

// ---------------------------------------------------------------------------
extern "C" void kernel_launch(void* const* d_in, const int* in_sizes, int n_in,
                              void* d_out, int out_size, void* d_ws, size_t ws_size,
                              hipStream_t stream) {
  (void)in_sizes; (void)n_in; (void)out_size; (void)ws_size;
  const int*   atom_types = (const int*)  d_in[0];
  const float* frac       = (const float*)d_in[1];
  /* d_in[2] lattice — unused by the forward math */
  const float* t_in       = (const float*)d_in[3];
  const float* props      = (const float*)d_in[4];
  const float* emb        = (const float*)d_in[5];
  const float* cW         = (const float*)d_in[6];
  const float* cb         = (const float*)d_in[7];
  const float* tW1        = (const float*)d_in[8];
  const float* tb1        = (const float*)d_in[9];
  const float* tW2        = (const float*)d_in[10];
  const float* tb2        = (const float*)d_in[11];
  const float* pW1        = (const float*)d_in[12];
  const float* pb1        = (const float*)d_in[13];
  const float* pW2        = (const float*)d_in[14];
  const float* pb2        = (const float*)d_in[15];
  const float* eW1hi      = (const float*)d_in[16];
  const float* eW1hj      = (const float*)d_in[17];
  const float* eW1d       = (const float*)d_in[18];
  const float* eb1        = (const float*)d_in[19];
  const float* eW2        = (const float*)d_in[20];
  const float* eb2        = (const float*)d_in[21];
  const float* nWh        = (const float*)d_in[22];
  const float* nWm        = (const float*)d_in[23];
  const float* nb1        = (const float*)d_in[24];
  const float* nW2        = (const float*)d_in[25];
  const float* nb2        = (const float*)d_in[26];
  const float* aW         = (const float*)d_in[27];
  const float* ab         = (const float*)d_in[28];
  const float* cvW1       = (const float*)d_in[29];
  const float* cvb1       = (const float*)d_in[30];
  const float* cvW2       = (const float*)d_in[31];
  const float* cvb2       = (const float*)d_in[32];
  const float* lvW1       = (const float*)d_in[33];
  const float* lvb1       = (const float*)d_in[34];
  const float* lvW2       = (const float*)d_in[35];
  const float* lvb2       = (const float*)d_in[36];

  float* out = (float*)d_out;

  // workspace layout (floats, then f16 tail)
  float* ws   = (float*)d_ws;
  float* h    = ws;              // 512*256 = 131072
  float* hi   = h    + 131072;   // 131072
  float* hj   = hi   + 131072;   // 131072
  float* msg  = hj   + 131072;   // 131072
  float* u    = msg  + 131072;   // 131072
  float* cond = u    + 131072;   // 1024
  float* dist = cond + 1024;     // 65536
  _Float16* W2T = (_Float16*)(dist + 65536);  // 6*256*256 f16 = 786KB (~3.6MB total)

  cvt_w2t<<<6 * 256, 256, 0, stream>>>(eW2, W2T);
  prelude_cond<<<4, 256, 0, stream>>>(t_in, props, tW1, tb1, tW2, tb2,
                                      pW1, pb1, pW2, pb2, cond);
  embed_h<<<512, 256, 0, stream>>>(atom_types, frac, emb, cW, cb, cond, h);
  dist_pbc<<<512, 128, 0, stream>>>(frac, dist);

  for (int l = 0; l < 6; ++l) {
    // hi = h @ eW1hi[l] ; hj = h @ eW1hj[l]
    gemm256_wmma<false, false, false, false><<<8, 256, 0, stream>>>(
        h, eW1hi + l * 65536, nullptr, nullptr, nullptr, nullptr, hi);
    gemm256_wmma<false, false, false, false><<<8, 256, 0, stream>>>(
        h, eW1hj + l * 65536, nullptr, nullptr, nullptr, nullptr, hj);

    // msg = sum_j silu( silu(hi_i + hj_j + dist*w + b) @ eW2 + eb2 )   (fused)
    edge_msg_wmma<<<512, 256, 0, stream>>>(hi, hj, dist, eW1d + l * 256,
                                           eb1 + l * 256, W2T + l * 65536,
                                           eb2 + l * 256, msg);

    // u = silu(h @ nWh + msg @ nWm + nb1) ;  h += u @ nW2 + nb2
    gemm256_wmma<true, true, false, true><<<8, 256, 0, stream>>>(
        h, nWh + l * 65536, msg, nWm + l * 65536, nb1 + l * 256, nullptr, u);
    gemm256_wmma<false, false, true, true><<<8, 256, 0, stream>>>(
        u, nW2 + l * 65536, nullptr, nullptr, nb2 + l * 256, h, h);
  }

  // outputs: atom_vf (51200) | coord_vf (1536) | lattice_vf (24)
  atom_head<<<512, 128, 0, stream>>>(h, aW, ab, out);
  coord_head<<<512, 256, 0, stream>>>(h, cvW1, cvb1, cvW2, cvb2, out + 51200);
  lattice_head<<<4, 256, 0, stream>>>(h, lvW1, lvb1, lvW2, lvb2, out + 51200 + 1536);
}
